// DLRANet_83811991814293
// MI455X (gfx1250) — compile-verified, matrix operand order checked by reading, not run on hardware
//
#include <hip/hip_runtime.h>
#include <hip/hip_bf16.h>
#include <math.h>

// ---------------------------------------------------------------------------
// DLRANet forward, low-rank path:  z = relu((z @ K) @ Vt) x3, logits, logsoftmax
// f16 WMMA (v_wmma_f32_16x16x32_f16) with f32 accumulate, wave32 / gfx1250.
// Round 2: 32-bit indexing + compile-time leading dims (kills v_mul_u64 chains),
//          split-K (grid.z=4) for the N=128 K-projection GEMMs + deterministic
//          reduce, smaller block tile for the skinny GEMMs (64 -> 512 blocks).
// ---------------------------------------------------------------------------

typedef __attribute__((ext_vector_type(8)))  _Float16 v8h;
typedef __attribute__((ext_vector_type(16))) _Float16 v16h;
typedef __attribute__((ext_vector_type(8)))  float    v8f;

#define B_ROWS 8192
#define D_IN   1024
#define WID    4096
#define D_OUT  1000
#define D_OUTP 1024   // padded logits width
#define RANK   128
#define SPLITS 4      // split-K factor for the N=128 GEMMs

// ---------------------------------------------------------------------------
// Fragment load: 16x32 f16 tile operand, per CDNA5 ISA 16-bit A layout.
// lane<16  holds row=lane,    K = {0..7, 16..23}
// lane>=16 holds row=lane-16, K = {8..15, 24..31}
// Two contiguous 16-byte chunks per lane -> two global_load_b128.
// B operand is fed from a transposed (N x K row-major) copy, same code path.
// ---------------------------------------------------------------------------
__device__ __forceinline__ v16h load_frag16(const _Float16* __restrict__ base,
                                            unsigned off, int ld) {
    const int lane = threadIdx.x & 31;
    const unsigned row = lane & 15;
    const unsigned kh  = (lane >> 4) << 3;          // 0 or 8
    const _Float16* p = base + (off + row * (unsigned)ld + kh);
    v8h lo = *(const v8h*)(p);
    v8h hi = *(const v8h*)(p + 16);
    return __builtin_shufflevector(lo, hi, 0,1,2,3,4,5,6,7,8,9,10,11,12,13,14,15);
}

// ---------------------------------------------------------------------------
// GEMM: C[M,N] (+=split) = A[M,K] * Bt[N,K]^T
//   A f16 row-major ld=LDK, Bt f16 row-major ld=LDK.
//   Wave tile = (WM_T*16) x (WN_T*16); block = WAVES_M x WAVES_N waves.
//   grid.z = split-K index; each split handles kCount K-elements and writes to
//   Cout + blockIdx.z * splitStride (f32 partials when splitting).
// ---------------------------------------------------------------------------
template <int WM_T, int WN_T, int WAVES_M, int WAVES_N,
          int LDK, int LDC, bool RELU, bool OUT_HALF>
__global__ __launch_bounds__(WAVES_M * WAVES_N * 32)
void gemm_wmma(const _Float16* __restrict__ A,
               const _Float16* __restrict__ Bt,
               void* __restrict__ Cout,
               int kCount, unsigned splitStride) {
    const int lane = threadIdx.x & 31;
    const int wave = threadIdx.x >> 5;
    const int wm   = wave % WAVES_M;
    const int wn   = wave / WAVES_M;
    const unsigned m_wave = blockIdx.y * (WAVES_M * WM_T * 16) + wm * (WM_T * 16);
    const unsigned n_wave = blockIdx.x * (WAVES_N * WN_T * 16) + wn * (WN_T * 16);
    const unsigned k0     = blockIdx.z * (unsigned)kCount;

    const unsigned aBase = m_wave * (unsigned)LDK + k0;
    const unsigned bBase = n_wave * (unsigned)LDK + k0;

    v8f acc[WM_T][WN_T];
#pragma unroll
    for (int i = 0; i < WM_T; ++i)
#pragma unroll
        for (int j = 0; j < WN_T; ++j)
            acc[i][j] = v8f{};

    for (int kt = 0; kt < kCount; kt += 32) {
        v16h a[WM_T], b[WN_T];
#pragma unroll
        for (int i = 0; i < WM_T; ++i)
            a[i] = load_frag16(A, aBase + (unsigned)(i * 16 * LDK + kt), LDK);
#pragma unroll
        for (int j = 0; j < WN_T; ++j)
            b[j] = load_frag16(Bt, bBase + (unsigned)(j * 16 * LDK + kt), LDK);
#pragma unroll
        for (int i = 0; i < WM_T; ++i)
#pragma unroll
            for (int j = 0; j < WN_T; ++j)
                acc[i][j] = __builtin_amdgcn_wmma_f32_16x16x32_f16(
                    /*neg_a=*/false, a[i], /*neg_b=*/false, b[j],
                    /*c_mod=*/(short)0, acc[i][j],
                    /*reuse_a=*/false, /*reuse_b=*/false);
    }

    // C/D layout: VGPR r -> row r (lanes 0-15) or row r+8 (lanes 16-31), col = lane&15
    const unsigned col   = lane & 15;
    const unsigned rbase = (lane >> 4) << 3;        // 0 or 8
    const unsigned zoff  = blockIdx.z * splitStride;
#pragma unroll
    for (int i = 0; i < WM_T; ++i) {
#pragma unroll
        for (int j = 0; j < WN_T; ++j) {
            const unsigned gn = n_wave + j * 16 + col;
#pragma unroll
            for (int r = 0; r < 8; ++r) {
                const unsigned gm = m_wave + i * 16 + rbase + r;
                float v = acc[i][j][r];
                if (RELU) v = fmaxf(v, 0.0f);
                if (OUT_HALF)
                    ((_Float16*)Cout)[zoff + gm * (unsigned)LDC + gn] = (_Float16)v;
                else
                    ((float*)Cout)[zoff + gm * (unsigned)LDC + gn] = v;
            }
        }
    }
}

// ---------------------------------------------------------------------------
// Deterministic 4-way split-K reduce + f32 -> f16 convert.
// ---------------------------------------------------------------------------
__global__ __launch_bounds__(256)
void reduce4_cvt(const float* __restrict__ P, _Float16* __restrict__ T, unsigned n) {
    unsigned i = blockIdx.x * 256u + threadIdx.x;
    unsigned stride = gridDim.x * 256u;
    for (; i < n; i += stride) {
        float s = (P[i] + P[i + n]) + (P[i + 2u * n] + P[i + 3u * n]);
        T[i] = (_Float16)s;
    }
}

// ---------------------------------------------------------------------------
// f32 -> f16 elementwise convert (grid-stride)
// ---------------------------------------------------------------------------
__global__ __launch_bounds__(256)
void cvt_f32_f16(const float* __restrict__ in, _Float16* __restrict__ out, unsigned n) {
    unsigned i = blockIdx.x * 256u + threadIdx.x;
    unsigned stride = gridDim.x * 256u;
    for (; i < n; i += stride) out[i] = (_Float16)in[i];
}

// ---------------------------------------------------------------------------
// Transpose + convert: out[c*rows + r] = (c < cols) ? in[r*cols + c] : 0
// out is cols_pad x rows, f16 row-major. Weights are tiny -> simple version.
// ---------------------------------------------------------------------------
__global__ __launch_bounds__(256)
void transpose_cvt(const float* __restrict__ in, _Float16* __restrict__ out,
                   int rows, int cols, int cols_pad) {
    unsigned idx = blockIdx.x * 256u + threadIdx.x;
    unsigned total = (unsigned)cols_pad * (unsigned)rows;
    if (idx >= total) return;
    unsigned c = idx / (unsigned)rows;
    unsigned r = idx % (unsigned)rows;
    float v = (c < (unsigned)cols) ? in[r * (unsigned)cols + c] : 0.0f;
    out[c * (unsigned)rows + r] = (_Float16)v;
}

// ---------------------------------------------------------------------------
// Row-wise log_softmax: one 256-thread block (8 waves, wave32) per row.
// ---------------------------------------------------------------------------
__global__ __launch_bounds__(256)
void logsoftmax_rows(const float* __restrict__ logits, float* __restrict__ out,
                     int ncols, int ldc) {
    const unsigned row = blockIdx.x;
    const float* in = logits + row * (unsigned)ldc;
    __shared__ float red[8];
    const int lane = threadIdx.x & 31;
    const int wid  = threadIdx.x >> 5;

    // --- max ---
    float m = -INFINITY;
    for (int c = threadIdx.x; c < ncols; c += blockDim.x) m = fmaxf(m, in[c]);
#pragma unroll
    for (int off = 16; off > 0; off >>= 1) m = fmaxf(m, __shfl_xor(m, off));
    if (lane == 0) red[wid] = m;
    __syncthreads();
    float bm = -INFINITY;
#pragma unroll
    for (int i = 0; i < 8; ++i) bm = fmaxf(bm, red[i]);
    __syncthreads();

    // --- sum exp ---
    float s = 0.0f;
    for (int c = threadIdx.x; c < ncols; c += blockDim.x) s += expf(in[c] - bm);
#pragma unroll
    for (int off = 16; off > 0; off >>= 1) s += __shfl_xor(s, off);
    if (lane == 0) red[wid] = s;
    __syncthreads();
    float tot = 0.0f;
#pragma unroll
    for (int i = 0; i < 8; ++i) tot += red[i];
    const float lse = bm + logf(tot);

    for (int c = threadIdx.x; c < ncols; c += blockDim.x)
        out[row * (unsigned)ncols + c] = in[c] - lse;
}

// ---------------------------------------------------------------------------
extern "C" void kernel_launch(void* const* d_in, const int* in_sizes, int n_in,
                              void* d_out, int out_size, void* d_ws, size_t ws_size,
                              hipStream_t stream) {
    (void)in_sizes; (void)n_in; (void)out_size; (void)ws_size;

    const float* x   = (const float*)d_in[0];
    const float* K0  = (const float*)d_in[1];
    const float* Vt0 = (const float*)d_in[2];
    const float* K1  = (const float*)d_in[3];
    const float* Vt1 = (const float*)d_in[4];
    const float* K2  = (const float*)d_in[5];
    const float* Vt2 = (const float*)d_in[6];
    const float* K3  = (const float*)d_in[7];
    const float* Vt3 = (const float*)d_in[8];

    char* ws = (char*)d_ws;
    const size_t MB = 1024u * 1024u;
    // Region A: z ping buffer (f16 8192x4096 = 64MB)
    _Float16* zA = (_Float16*)(ws);
    // Region B (64MB): xh (16MB, dead after layer0 GEMM1), then zB, then logits(32MB)
    _Float16* xh     = (_Float16*)(ws + 64 * MB);
    _Float16* zB     = (_Float16*)(ws + 64 * MB);
    float*    logits = (float*)   (ws + 64 * MB);
    _Float16* T      = (_Float16*)(ws + 128 * MB);   // 8192x128 f16 = 2MB
    char* w = ws + 130 * MB;                         // transposed f16 weights (~6.5MB)
    _Float16* Kt0  = (_Float16*)(w);                 w += (size_t)RANK * D_IN * 2;
    _Float16* Kt1  = (_Float16*)(w);                 w += (size_t)RANK * WID * 2;
    _Float16* Kt2  = (_Float16*)(w);                 w += (size_t)RANK * WID * 2;
    _Float16* Kt3  = (_Float16*)(w);                 w += (size_t)RANK * WID * 2;
    _Float16* Vt0t = (_Float16*)(w);                 w += (size_t)WID * RANK * 2;
    _Float16* Vt1t = (_Float16*)(w);                 w += (size_t)WID * RANK * 2;
    _Float16* Vt2t = (_Float16*)(w);                 w += (size_t)WID * RANK * 2;
    _Float16* Vt3t = (_Float16*)(w);
    float* Pk = (float*)(ws + 140 * MB);             // split-K partials: 4 x 8192x128 f32 = 16MB

    const unsigned TN = (unsigned)B_ROWS * RANK;     // elements in T / one partial slab

    // ---- prep: convert x, transpose+convert all weights ----
    {
        cvt_f32_f16<<<4096, 256, 0, stream>>>(x, xh, (unsigned)B_ROWS * D_IN);
        auto tlaunch = [&](const float* in, _Float16* outp, int rows, int cols, int cols_pad) {
            unsigned tot = (unsigned)cols_pad * rows;
            transpose_cvt<<<(tot + 255) / 256, 256, 0, stream>>>(in, outp, rows, cols, cols_pad);
        };
        tlaunch(K0,  Kt0,  D_IN, RANK, RANK);    // [1024,128] -> [128,1024]
        tlaunch(K1,  Kt1,  WID,  RANK, RANK);    // [4096,128] -> [128,4096]
        tlaunch(K2,  Kt2,  WID,  RANK, RANK);
        tlaunch(K3,  Kt3,  WID,  RANK, RANK);
        tlaunch(Vt0, Vt0t, RANK, WID,  WID);     // [128,4096] -> [4096,128]
        tlaunch(Vt1, Vt1t, RANK, WID,  WID);
        tlaunch(Vt2, Vt2t, RANK, WID,  WID);
        tlaunch(Vt3, Vt3t, RANK, D_OUT, D_OUTP); // [128,1000] -> [1024,128] zero-padded
    }

    dim3 blk(256);

    // ---- K-projection (split-K): P[z] = A * Kt (partial f32), then reduce -> T f16
    //   wave tile 32x32, block 64x128, grid (1, 128, 4) = 512 blocks
    auto proj = [&](const _Float16* Ain, const _Float16* Ktw, auto ldkTag) {
        constexpr int LDKv = decltype(ldkTag)::value;
        dim3 g(RANK / 128, B_ROWS / 64, SPLITS);
        gemm_wmma<2, 2, 2, 4, LDKv, RANK, false, false>
            <<<g, blk, 0, stream>>>(Ain, Ktw, (void*)Pk, LDKv / SPLITS, TN);
        reduce4_cvt<<<2048, 256, 0, stream>>>(Pk, T, TN);
    };
    // ---- expansion: z' = act(T @ Vt^T) -- wave tile 64x32, block 128x128
    auto expand = [&](const _Float16* Vtt, void* outp, auto reluTag, auto halfTag, auto ldcTag) {
        constexpr bool RELUv = decltype(reluTag)::value;
        constexpr bool HALFv = decltype(halfTag)::value;
        constexpr int  LDCv  = decltype(ldcTag)::value;
        dim3 g(LDCv / 128, B_ROWS / 128, 1);
        gemm_wmma<4, 2, 2, 4, RANK, LDCv, RELUv, HALFv>
            <<<g, blk, 0, stream>>>(T, Vtt, outp, RANK, 0u);
    };

    using TrueT  = std::integral_constant<bool, true>;
    using FalseT = std::integral_constant<bool, false>;
    using LdDin  = std::integral_constant<int, D_IN>;
    using LdWid  = std::integral_constant<int, WID>;
    using LdOutP = std::integral_constant<int, D_OUTP>;

    // ---- layer 0 ----
    proj(xh, Kt0, LdDin{});
    expand(Vt0t, (void*)zA, TrueT{}, TrueT{}, LdWid{});
    // ---- layer 1 ----
    proj(zA, Kt1, LdWid{});
    expand(Vt1t, (void*)zB, TrueT{}, TrueT{}, LdWid{});
    // ---- layer 2 ----
    proj(zB, Kt2, LdWid{});
    expand(Vt2t, (void*)zA, TrueT{}, TrueT{}, LdWid{});
    // ---- layer 3 (logits, f32, padded N=1024; padded cols are exactly 0) ----
    proj(zA, Kt3, LdWid{});
    expand(Vt3t, (void*)logits, FalseT{}, FalseT{}, LdOutP{});

    // ---- log_softmax over the first 1000 columns ----
    logsoftmax_rows<<<B_ROWS, 256, 0, stream>>>(logits, (float*)d_out, D_OUT, D_OUTP);
}
